// MFD_28381143892634
// MI455X (gfx1250) — compile-verified, matrix operand order checked by reading, not run on hardware
//
#include <hip/hip_runtime.h>
#include <hip/hip_bf16.h>

// ---------------------------------------------------------------------------
// Cross-attention for MI455X (gfx1250, wave32, WMMA).
//   B=2, N1=N2=2048, H=768, heads=12, d=64.
// Pipeline:
//   1) cvt_f32_f16 : s1,s2 -> f16
//   2) transpose_w : W[k][n] fp32 -> Wt[n][k] f16  (3 mats)
//   3) proj_kernel : WMMA GEMM (32x32 tile/wave) -> Qh[b,h,n,64], Kh[b,h,n,64],
//                    Vt[b,h,64,n] f16 (V transposed for P*V B-fragments)
//   4) attn_kernel : block = 12 waves = 12 heads of one (b, 16-query tile).
//                    2-pass flash softmax per wave (WMMA QK^T, WMMA P*V);
//                    cross-head prob sum via LDS ds_add_f32 + coalesced store.
// ---------------------------------------------------------------------------

typedef _Float16 v16h __attribute__((ext_vector_type(16)));
typedef _Float16 v8h  __attribute__((ext_vector_type(8)));
typedef float    v8f  __attribute__((ext_vector_type(8)));

#define HIDDEN   768
#define HEADS    12
#define HEADDIM  64
#define SEQ      2048
#define BATCH    2

// ---------------------------------------------------------------------------
// Fragment loader for 16x32 f16 A-matrix (row=M, cols=K) OR 32x16 B-matrix
// (lane r holds column N=r, K elements along rows of a K-contiguous buffer).
// Per CDNA5 ISA 7.12.2: lanes 0-15 hold K={0..7,16..23}, lanes 16-31 hold
// K={8..15,24..31} for row M = lane%16. Two 16-byte loads per lane.
// ---------------------------------------------------------------------------
__device__ __forceinline__ v16h load_frag16x32(const _Float16* p, int ld) {
    const int lane = threadIdx.x & 31;
    const int r    = lane & 15;
    const int hf   = lane >> 4;
    const _Float16* q = p + (size_t)r * ld + hf * 8;
    v8h lo = *(const v8h*)(q);
    v8h hi = *(const v8h*)(q + 16);
    return __builtin_shufflevector(lo, hi, 0,1,2,3,4,5,6,7,8,9,10,11,12,13,14,15);
}

__device__ __forceinline__ v8f wmma_f16(v16h a, v16h b, v8f c) {
    return __builtin_amdgcn_wmma_f32_16x16x32_f16(
        /*neg_a=*/false, a, /*neg_b=*/false, b,
        /*c_mod=*/(short)0, c, /*reuse_a=*/false, /*reuse_b=*/false);
}

// reductions across the 16-lane half (C-matrix row lives in 16 lanes)
__device__ __forceinline__ float rowmax16(float v) {
    v = fmaxf(v, __shfl_xor(v, 1, 32));
    v = fmaxf(v, __shfl_xor(v, 2, 32));
    v = fmaxf(v, __shfl_xor(v, 4, 32));
    v = fmaxf(v, __shfl_xor(v, 8, 32));
    return v;
}
__device__ __forceinline__ float rowsum16(float v) {
    v += __shfl_xor(v, 1, 32);
    v += __shfl_xor(v, 2, 32);
    v += __shfl_xor(v, 4, 32);
    v += __shfl_xor(v, 8, 32);
    return v;
}

// ---------------------------------------------------------------------------
// 1) fp32 -> f16 convert (grid-stride)
// ---------------------------------------------------------------------------
__global__ void cvt_f32_f16(const float* __restrict__ in, _Float16* __restrict__ out, int n) {
    for (int i = blockIdx.x * blockDim.x + threadIdx.x; i < n; i += gridDim.x * blockDim.x)
        out[i] = (_Float16)in[i];
}

// ---------------------------------------------------------------------------
// 2) transpose W[k][n] -> Wt[mat][n][k] (f16, k contiguous)
// ---------------------------------------------------------------------------
__global__ void transpose_w(const float* __restrict__ Wq, const float* __restrict__ Wk,
                            const float* __restrict__ Wv, _Float16* __restrict__ Wt) {
    const int per = HIDDEN * HIDDEN;
    for (int i = blockIdx.x * blockDim.x + threadIdx.x; i < 3 * per;
         i += gridDim.x * blockDim.x) {
        int mat = i / per, rem = i % per;
        int n = rem / HIDDEN, k = rem % HIDDEN;
        const float* W = (mat == 0) ? Wq : (mat == 1) ? Wk : Wv;
        Wt[i] = (_Float16)W[(size_t)k * HIDDEN + n];
    }
}

// ---------------------------------------------------------------------------
// 3) QKV projection: one wave per 32x32 output tile (2x2 WMMA, frag reuse).
//    blockDim=128 (4 waves), grid=(24/4, 4096/32, 3 mats)
// ---------------------------------------------------------------------------
__global__ void proj_kernel(const _Float16* __restrict__ Xh1, const _Float16* __restrict__ Xh2,
                            const _Float16* __restrict__ Wt,
                            const float* __restrict__ bq, const float* __restrict__ bk,
                            const float* __restrict__ bv,
                            _Float16* __restrict__ Qh, _Float16* __restrict__ Kh,
                            _Float16* __restrict__ Vt) {
    const int wave = threadIdx.x >> 5;
    const int lane = threadIdx.x & 31;
    const int r = lane & 15, hf = lane >> 4;
    const int mat = blockIdx.z;
    const int n0  = (blockIdx.x * 4 + wave) * 32;
    const int m0  = blockIdx.y * 32;

    const _Float16* X = (mat == 0) ? Xh1 : Xh2;
    const _Float16* W = Wt + (size_t)mat * HIDDEN * HIDDEN;
    const float* bias = (mat == 0) ? bq : (mat == 1) ? bk : bv;

    v8f acc[2][2] = {{{}, {}}, {{}, {}}};
    for (int k0 = 0; k0 < HIDDEN; k0 += 32) {
        v16h a0 = load_frag16x32(X + (size_t)m0 * HIDDEN + k0, HIDDEN);
        v16h a1 = load_frag16x32(X + (size_t)(m0 + 16) * HIDDEN + k0, HIDDEN);
        v16h b0 = load_frag16x32(W + (size_t)n0 * HIDDEN + k0, HIDDEN);
        v16h b1 = load_frag16x32(W + (size_t)(n0 + 16) * HIDDEN + k0, HIDDEN);
        acc[0][0] = wmma_f16(a0, b0, acc[0][0]);
        acc[0][1] = wmma_f16(a0, b1, acc[0][1]);
        acc[1][0] = wmma_f16(a1, b0, acc[1][0]);
        acc[1][1] = wmma_f16(a1, b1, acc[1][1]);
    }
#pragma unroll
    for (int ni = 0; ni < 2; ++ni) {
        const int nn   = n0 + 16 * ni;
        const float bn = bias[nn + r];
        const int head = nn / HEADDIM;
        const int dd   = (nn % HEADDIM) + r;
#pragma unroll
        for (int mi = 0; mi < 2; ++mi) {
#pragma unroll
            for (int j = 0; j < 8; ++j) {
                int row = m0 + 16 * mi + j + 8 * hf;   // row in flattened [B*SEQ]
                int bb  = row >> 11;                   // /2048
                int nr  = row & (SEQ - 1);
                _Float16 hv = (_Float16)(acc[mi][ni][j] + bn);
                size_t hbase = (size_t)(bb * HEADS + head);
                if (mat == 0)      Qh[(hbase * SEQ + nr) * HEADDIM + dd] = hv;
                else if (mat == 1) Kh[(hbase * SEQ + nr) * HEADDIM + dd] = hv;
                else               Vt[(hbase * HEADDIM + dd) * SEQ + nr] = hv;
            }
        }
    }
}

// ---------------------------------------------------------------------------
// 4) attention: block = 12 waves (one per head) for one (b, 16-query tile).
//    Cross-head prob sum accumulated in LDS (ds_add_f32), stored coalesced.
// ---------------------------------------------------------------------------
__global__ __launch_bounds__(384) void
attn_kernel(const _Float16* __restrict__ Qh, const _Float16* __restrict__ Kh,
            const _Float16* __restrict__ Vt, const float* __restrict__ mask,
            float* __restrict__ ctx, float* __restrict__ attnw) {
    __shared__ __align__(16) _Float16 pBuf[HEADS][16][40]; // per-wave 16q x 32k probs
    __shared__ __align__(16) float    pSum[16][32];        // cross-head sum tile

    const int head = threadIdx.x >> 5;                     // wave id == head
    const int lane = threadIdx.x & 31;
    const int r = lane & 15, hf = lane >> 4;

    const int b     = blockIdx.x / (SEQ / 16);
    const int qbase = (blockIdx.x % (SEQ / 16)) * 16;

    const _Float16* Qp = Qh + ((size_t)(b * HEADS + head) * SEQ + qbase) * HEADDIM;
    const _Float16* Kp = Kh + (size_t)(b * HEADS + head) * SEQ * HEADDIM;
    const _Float16* Vp = Vt + (size_t)(b * HEADS + head) * HEADDIM * SEQ;
    const float*    mp = mask + (size_t)b * SEQ;

    const v16h Aq0 = load_frag16x32(Qp + 0,  HEADDIM);
    const v16h Aq1 = load_frag16x32(Qp + 32, HEADDIM);

    const float scale = 0.125f; // 1/sqrt(64)
    float m[8], l[8];
#pragma unroll
    for (int j = 0; j < 8; ++j) { m[j] = -3.0e38f; l[j] = 0.0f; }

    // ---- pass 1: streaming row max / sum-of-exp over all keys ----
    for (int kb = 0; kb < SEQ; kb += 16) {
        if (kb + 16 < SEQ)
            __builtin_prefetch(Kp + (size_t)(kb + 16) * HEADDIM, 0, 1);
        v8f s = {};
        s = wmma_f16(Aq0, load_frag16x32(Kp + (size_t)kb * HEADDIM + 0,  HEADDIM), s);
        s = wmma_f16(Aq1, load_frag16x32(Kp + (size_t)kb * HEADDIM + 32, HEADDIM), s);
        const float mk = mp[kb + r];
#pragma unroll
        for (int j = 0; j < 8; ++j) {
            float z  = s[j] * scale + mk;
            float nm = fmaxf(m[j], rowmax16(z));
            float p  = __expf(z - nm);
            float su = rowsum16(p);
            l[j] = l[j] * __expf(m[j] - nm) + su;
            m[j] = nm;
        }
    }
    float linv[8];
#pragma unroll
    for (int j = 0; j < 8; ++j) linv[j] = 1.0f / l[j];

    // ---- pass 2: probs -> LDS head-sum + P*V WMMA (lockstep over key blocks)
    v8f acc[4] = {{}, {}, {}, {}};
    for (int kb = 0; kb < SEQ; kb += 32) {
        __syncthreads();                       // prev block's pSum reads done
        for (int i = threadIdx.x; i < 16 * 32; i += 384)
            (&pSum[0][0])[i] = 0.0f;
        __syncthreads();                       // pSum cleared

#pragma unroll
        for (int t = 0; t < 2; ++t) {
            v8f s = {};
            s = wmma_f16(Aq0, load_frag16x32(Kp + (size_t)(kb + 16 * t) * HEADDIM + 0,  HEADDIM), s);
            s = wmma_f16(Aq1, load_frag16x32(Kp + (size_t)(kb + 16 * t) * HEADDIM + 32, HEADDIM), s);
            const float mk = mp[kb + 16 * t + r];
#pragma unroll
            for (int j = 0; j < 8; ++j) {
                float z = s[j] * scale + mk;
                float p = __expf(z - m[j]) * linv[j];
                atomicAdd(&pSum[j + 8 * hf][16 * t + r], p * (1.0f / (float)HEADS));
                pBuf[head][j + 8 * hf][16 * t + r] = (_Float16)p;
            }
        }
        // LDS in-order per wave; wait before reading own prob tile back
        asm volatile("s_wait_dscnt 0" ::: "memory");
        const v16h Pf = load_frag16x32(&pBuf[head][0][0], 40);
#pragma unroll
        for (int c = 0; c < 4; ++c)
            acc[c] = wmma_f16(Pf, load_frag16x32(Vp + (size_t)(c * 16) * SEQ + kb, SEQ), acc[c]);

        __syncthreads();                       // all heads accumulated into pSum
        for (int i = threadIdx.x; i < 16 * 32; i += 384) {
            int row = i >> 5, col = i & 31;
            attnw[((size_t)b * SEQ + qbase + row) * SEQ + kb + col] = pSum[row][col];
        }
    }

    // ---- write ctx [b, q, head*64 + d] ----
#pragma unroll
    for (int c = 0; c < 4; ++c) {
#pragma unroll
        for (int j = 0; j < 8; ++j) {
            int q   = qbase + j + 8 * hf;
            int col = head * HEADDIM + c * 16 + r;
            ctx[((size_t)b * SEQ + q) * HIDDEN + col] = acc[c][j];
        }
    }
}

// ---------------------------------------------------------------------------
extern "C" void kernel_launch(void* const* d_in, const int* in_sizes, int n_in,
                              void* d_out, int out_size, void* d_ws, size_t ws_size,
                              hipStream_t stream) {
    const float* s1   = (const float*)d_in[0];
    const float* s2   = (const float*)d_in[1];
    const float* mask = (const float*)d_in[2];
    const float* Wq   = (const float*)d_in[3];
    const float* bq   = (const float*)d_in[4];
    const float* Wk   = (const float*)d_in[5];
    const float* bk   = (const float*)d_in[6];
    const float* Wv   = (const float*)d_in[7];
    const float* bv   = (const float*)d_in[8];

    const int NX = BATCH * SEQ * HIDDEN;      // 3,145,728 per activation matrix
    const int NW = 3 * HIDDEN * HIDDEN;       // 1,769,472

    _Float16* Xh1 = (_Float16*)d_ws;
    _Float16* Xh2 = Xh1 + NX;
    _Float16* Wt  = Xh2 + NX;
    _Float16* Qh  = Wt  + NW;
    _Float16* Kh  = Qh  + NX;
    _Float16* Vt  = Kh  + NX;

    float* ctx   = (float*)d_out;
    float* attnw = ctx + (size_t)NX;          // ctx is fp32, same elem count

    cvt_f32_f16<<<2048, 256, 0, stream>>>(s1, Xh1, NX);
    cvt_f32_f16<<<2048, 256, 0, stream>>>(s2, Xh2, NX);
    transpose_w<<<2048, 256, 0, stream>>>(Wq, Wk, Wv, Wt);

    // 24 n-tiles / 4 waves-per-block, 128 m-tiles, 3 matrices
    proj_kernel<<<dim3(HIDDEN / 32 / 4, (BATCH * SEQ) / 32, 3), 128, 0, stream>>>(
        Xh1, Xh2, Wt, bq, bk, bv, Qh, Kh, Vt);

    // one block (12 waves = 12 heads) per (b, 16-query tile): 256 blocks
    attn_kernel<<<BATCH * (SEQ / 16), 384, 0, stream>>>(
        Qh, Kh, Vt, mask, ctx, attnw);
}
